// WordGraphModel_1726576853058
// MI455X (gfx1250) — matrix operand order repaired and checked by reading, not compiled, not run on hardware
//
#include <hip/hip_runtime.h>
#include <hip/hip_bf16.h>
#include <math.h>

// Problem constants (match reference)
#define NNODE 16384
#define NEDGE 524288
#define NGRAPH 64
#define DIN   300
#define DINP  320   // DIN padded to multiple of 32 for WMMA K-loop
#define DH    256
#define NLAYER 3
#define DOUT  4

typedef __attribute__((ext_vector_type(16))) _Float16 v16h;
typedef __attribute__((ext_vector_type(8)))  _Float16 v8h;
typedef __attribute__((ext_vector_type(8)))  float    v8f;

#define NEG_INF (-__builtin_huge_valf())

__device__ __forceinline__ float wgm_sigmoid(float x) { return 1.0f / (1.0f + __expf(-x)); }

// float atomic max via int/uint monotone-encoding trick
__device__ __forceinline__ float wgm_atomic_max_f32(float* addr, float val) {
  if (val >= 0.0f)
    return __int_as_float(atomicMax((int*)addr, __float_as_int(val)));
  else
    return __uint_as_float(atomicMin((unsigned int*)addr, __float_as_uint(val)));
}

// ---------------------------------------------------------------------------
// f32 -> f16 conversion with optional column padding (k0 valid cols -> k1)
// ---------------------------------------------------------------------------
__global__ void wgm_cvt_f16(const float* __restrict__ src, _Float16* __restrict__ dst,
                            int rows, int k0, int k1) {
  size_t i = (size_t)blockIdx.x * blockDim.x + threadIdx.x;
  if (i >= (size_t)rows * k1) return;
  int r = (int)(i / k1);
  int c = (int)(i % k1);
  dst[i] = (c < k0) ? (_Float16)src[(size_t)r * k0 + c] : (_Float16)0.0f;
}

__global__ void wgm_fill(float* __restrict__ p, float v, size_t n) {
  size_t i = (size_t)blockIdx.x * blockDim.x + threadIdx.x;
  if (i < n) p[i] = v;
}

// ---------------------------------------------------------------------------
// WMMA fragment helpers (CDNA5 ISA 7.12.2 layouts, wave32)
// A 16x32 f16 fragment: lane half 0 -> K {0..7,16..23}, half 1 -> {8..15,24..31}
// B 32x16 fragment: column n = lane&15, same K order as A: half 0 -> {0..7,16..23}, half 1 -> {8..15,24..31}
// ---------------------------------------------------------------------------
__device__ __forceinline__ v16h wgm_frag_a(const _Float16* row, int k0, int hf) {
  v8h lo = *(const v8h*)(row + k0 + hf * 8);
  v8h hi = *(const v8h*)(row + k0 + 16 + hf * 8);
  return __builtin_shufflevector(lo, hi, 0,1,2,3,4,5,6,7,8,9,10,11,12,13,14,15);
}
__device__ __forceinline__ v16h wgm_frag_b(const _Float16* row, int k0, int hf) {
  v8h lo = *(const v8h*)(row + k0 + hf * 8);
  v8h hi = *(const v8h*)(row + k0 + 16 + hf * 8);
  return __builtin_shufflevector(lo, hi, 0,1,2,3,4,5,6,7,8,9,10,11,12,13,14,15);
}
#define WGM_WMMA(a, b, c) \
  __builtin_amdgcn_wmma_f32_16x16x32_f16(false, (a), false, (b), (short)0, (c), false, false)

// ---------------------------------------------------------------------------
// WMMA GEMM: C[M,Nc] = A[M,K](f16) x B[Nc,K](f16)^T + bias[Nc]
// 256 threads = 8 waves arranged 4(M) x 2(N); each wave computes a 32x32 C
// tile with 2x2 register blocking (4 v8f accumulators) -> 1 fragment load
// per WMMA instead of 2. Block tile 128x64. Grids divide M/Nc exactly and the
// tile guard is wave-uniform, so EXEC is all-ones for every WMMA.
// ---------------------------------------------------------------------------
__global__ void __launch_bounds__(256)
wgm_gemm_wmma(const _Float16* __restrict__ A, const _Float16* __restrict__ B,
              const float* __restrict__ bias, float* __restrict__ C,
              int M, int Nc, int K) {
  const int wave = threadIdx.x >> 5;
  const int lane = threadIdx.x & 31;
  const int hf   = lane >> 4;   // which 16-lane half of the wave
  const int l16  = lane & 15;
  const int tileM = blockIdx.x * 128 + (wave & 3) * 32;
  const int tileN = blockIdx.y * 64 + (wave >> 2) * 32;
  if (tileM >= M || tileN >= Nc) return;  // uniform per wave

  const _Float16* arow0 = A + (size_t)(tileM + l16) * K;
  const _Float16* arow1 = arow0 + (size_t)16 * K;
  const _Float16* brow0 = B + (size_t)(tileN + l16) * K;
  const _Float16* brow1 = brow0 + (size_t)16 * K;

  v8f acc00 = {}, acc01 = {}, acc10 = {}, acc11 = {};
  for (int k0 = 0; k0 < K; k0 += 32) {
    // prefetch next K-chunk of the streamed A operand (global_prefetch_b8;
    // speculative: OOB translations are silently dropped)
    __builtin_prefetch(arow0 + k0 + 64, 0, 1);
    __builtin_prefetch(arow1 + k0 + 64, 0, 1);
    v16h a0 = wgm_frag_a(arow0, k0, hf);
    v16h a1 = wgm_frag_a(arow1, k0, hf);
    v16h b0 = wgm_frag_b(brow0, k0, hf);
    v16h b1 = wgm_frag_b(brow1, k0, hf);
    acc00 = WGM_WMMA(a0, b0, acc00);
    acc01 = WGM_WMMA(a0, b1, acc01);
    acc10 = WGM_WMMA(a1, b0, acc10);
    acc11 = WGM_WMMA(a1, b1, acc11);
  }

  const int n0 = tileN + l16;
  const int n1 = tileN + 16 + l16;
  const float bv0 = bias ? bias[n0] : 0.0f;
  const float bv1 = bias ? bias[n1] : 0.0f;
#pragma unroll
  for (int r = 0; r < 8; ++r) {
    int m0 = tileM + r + 8 * hf;        // C layout: VGPR r -> row r (+8 upper half)
    int m1 = m0 + 16;
    C[(size_t)m0 * Nc + n0] = acc00[r] + bv0;
    C[(size_t)m0 * Nc + n1] = acc01[r] + bv1;
    C[(size_t)m1 * Nc + n0] = acc10[r] + bv0;
    C[(size_t)m1 * Nc + n1] = acc11[r] + bv1;
  }
}

// ---------------------------------------------------------------------------
// Sequential GRU scan (the serial bottleneck): one persistent workgroup,
// h in LDS, 768 threads each own one gate row of Whh (L2-resident, 768 KB).
// gx already holds x @ Wih^T + bih (computed with WMMA).
// ---------------------------------------------------------------------------
__global__ void __launch_bounds__(768)
wgm_gru_scan(const float* __restrict__ gx, const float* __restrict__ Whh,
             const float* __restrict__ bhh, float* __restrict__ ys) {
  __shared__ float h[DH];
  __shared__ float gh[3 * DH];
  const int t = threadIdx.x;  // 0..767
  if (t < DH) h[t] = 0.0f;
  const float4* wrow = (const float4*)(Whh + (size_t)t * DH);
  const float bh = bhh[t];
  __syncthreads();
  for (int i = 0; i < NNODE; ++i) {
    float acc = 0.0f;
#pragma unroll 8
    for (int k = 0; k < DH / 4; ++k) {
      float4 w = wrow[k];
      acc += w.x * h[4 * k]     + w.y * h[4 * k + 1]
           + w.z * h[4 * k + 2] + w.w * h[4 * k + 3];
    }
    gh[t] = acc + bh;
    __syncthreads();
    if (t < DH) {
      const float* g = gx + (size_t)i * (3 * DH);
      float r  = wgm_sigmoid(g[t]          + gh[t]);
      float z  = wgm_sigmoid(g[DH + t]     + gh[DH + t]);
      float nn = tanhf     (g[2 * DH + t] + r * gh[2 * DH + t]);
      float hn = (1.0f - z) * nn + z * h[t];
      h[t] = hn;
      ys[(size_t)i * DH + t] = hn;
    }
    __syncthreads();
  }
}

// SwiGLU epilogue: h = tw * silu(tv)
__global__ void wgm_swiglu(const float* __restrict__ tw, const float* __restrict__ tv,
                           float* __restrict__ h, size_t n) {
  size_t i = (size_t)blockIdx.x * blockDim.x + threadIdx.x;
  if (i >= n) return;
  float v = tv[i];
  h[i] = tw[i] * (v * wgm_sigmoid(v));
}

// ---------------------------------------------------------------------------
// Attention: per-node dot products with att_src / att_dst (one wave per node)
// ---------------------------------------------------------------------------
__global__ void __launch_bounds__(256)
wgm_attn_dots(const float* __restrict__ h, const float* __restrict__ as,
              const float* __restrict__ ad, float* __restrict__ s, float* __restrict__ d) {
  const int node = blockIdx.x * 8 + (threadIdx.x >> 5);
  const int lane = threadIdx.x & 31;
  const float* hr = h + (size_t)node * DH;
  float accs = 0.0f, accd = 0.0f;
#pragma unroll
  for (int k = lane; k < DH; k += 32) {
    float hv = hr[k];
    accs += hv * as[k];
    accd += hv * ad[k];
  }
#pragma unroll
  for (int off = 16; off > 0; off >>= 1) {
    accs += __shfl_down(accs, off, 32);
    accd += __shfl_down(accd, off, 32);
  }
  if (lane == 0) { s[node] = accs; d[node] = accd; }
}

__global__ void wgm_edge_logit_max(const int* __restrict__ ei, const float* __restrict__ s,
                                   const float* __restrict__ d, float* __restrict__ logits,
                                   float* __restrict__ m) {
  int e = blockIdx.x * blockDim.x + threadIdx.x;
  if (e >= NEDGE) return;
  int src = ei[e], dst = ei[NEDGE + e];
  float x = s[src] + d[dst];
  float lg = (x > 0.0f) ? x : 0.2f * x;  // leaky_relu(0.2)
  logits[e] = lg;
  wgm_atomic_max_f32(&m[dst], lg);
}

__global__ void wgm_edge_exp_den(const int* __restrict__ ei, const float* __restrict__ logits,
                                 const float* __restrict__ m, float* __restrict__ eexp,
                                 float* __restrict__ den) {
  int e = blockIdx.x * blockDim.x + threadIdx.x;
  if (e >= NEDGE) return;
  int dst = ei[NEDGE + e];
  float ev = __expf(logits[e] - m[dst]);
  eexp[e] = ev;
  atomicAdd(&den[dst], ev);
}

// attended[dst,:] += alpha * h[src,:]  (8 edges per block, one wave per edge)
__global__ void __launch_bounds__(256)
wgm_edge_attend(const int* __restrict__ ei, const float* __restrict__ ew,
                const float* __restrict__ eexp, const float* __restrict__ den,
                const float* __restrict__ h, float* __restrict__ attended) {
  const int e = blockIdx.x * 8 + (threadIdx.x >> 5);
  const int lane = threadIdx.x & 31;
  if (e >= NEDGE) return;
  int src = ei[e], dst = ei[NEDGE + e];
  float alpha = eexp[e] / (den[dst] + 1e-9f) * ew[e];
  const float* hs = h + (size_t)src * DH;
  float* ad = attended + (size_t)dst * DH;
#pragma unroll
  for (int d0 = 0; d0 < DH; d0 += 32)
    atomicAdd(&ad[d0 + lane], alpha * hs[d0 + lane]);
}

// agg[dst,:] += ew * g[src,:] ; deg[dst] += ew
__global__ void __launch_bounds__(256)
wgm_edge_gnn(const int* __restrict__ ei, const float* __restrict__ ew,
             const float* __restrict__ g, float* __restrict__ agg, float* __restrict__ deg) {
  const int e = blockIdx.x * 8 + (threadIdx.x >> 5);
  const int lane = threadIdx.x & 31;
  if (e >= NEDGE) return;
  int src = ei[e], dst = ei[NEDGE + e];
  float w = ew[e];
  if (lane == 0) atomicAdd(&deg[dst], w);
  const float* gs = g + (size_t)src * DH;
  float* ag = agg + (size_t)dst * DH;
#pragma unroll
  for (int d0 = 0; d0 < DH; d0 += 32)
    atomicAdd(&ag[d0 + lane], w * gs[d0 + lane]);
}

// h = relu(agg / (deg + 1e-6)) + attended
__global__ void wgm_combine(const float* __restrict__ agg, const float* __restrict__ deg,
                            const float* __restrict__ att, float* __restrict__ h, size_t n) {
  size_t i = (size_t)blockIdx.x * blockDim.x + threadIdx.x;
  if (i >= n) return;
  int node = (int)(i / DH);
  float v = agg[i] / (deg[node] + 1e-6f);
  h[i] = fmaxf(v, 0.0f) + att[i];
}

// ---------------------------------------------------------------------------
// Readout: per-graph mean/max pool + final tiny GEMM
// ---------------------------------------------------------------------------
__global__ void __launch_bounds__(256)
wgm_pool(const float* __restrict__ h, const int* __restrict__ batch,
         float* __restrict__ sum, float* __restrict__ mx, float* __restrict__ cnt) {
  const int node = blockIdx.x * 8 + (threadIdx.x >> 5);
  const int lane = threadIdx.x & 31;
  const int b = batch[node];
  if (lane == 0) atomicAdd(&cnt[b], 1.0f);
#pragma unroll
  for (int d0 = 0; d0 < DH; d0 += 32) {
    float v = h[(size_t)node * DH + d0 + lane];
    atomicAdd(&sum[(size_t)b * DH + d0 + lane], v);
    wgm_atomic_max_f32(&mx[(size_t)b * DH + d0 + lane], v);
  }
}

__global__ void wgm_readout(const float* __restrict__ sum, const float* __restrict__ mx,
                            const float* __restrict__ cnt, const float* __restrict__ out_W,
                            const float* __restrict__ out_b, float* __restrict__ out) {
  int idx = blockIdx.x * blockDim.x + threadIdx.x;
  if (idx >= NGRAPH * DOUT) return;
  int g = idx / DOUT, o = idx % DOUT;
  float c = fmaxf(cnt[g], 1.0f);
  float acc = out_b[o];
  for (int k = 0; k < DH; ++k) {
    float mean = sum[(size_t)g * DH + k] / c;
    float mval = mx[(size_t)g * DH + k];
    if (mval < -3.0e38f) mval = 0.0f;  // empty-graph -inf -> 0
    acc += out_W[o * 2 * DH + k] * mean + out_W[o * 2 * DH + DH + k] * mval;
  }
  out[g * DOUT + o] = acc;
}

// ---------------------------------------------------------------------------
extern "C" void kernel_launch(void* const* d_in, const int* in_sizes, int n_in,
                              void* d_out, int out_size, void* d_ws, size_t ws_size,
                              hipStream_t stream) {
  (void)in_sizes; (void)n_in; (void)out_size; (void)ws_size;
  const float* x       = (const float*)d_in[0];
  const int*   ei      = (const int*)d_in[1];
  const float* ew      = (const float*)d_in[2];
  const int*   batch   = (const int*)d_in[3];
  const float* in_W    = (const float*)d_in[4];
  const float* in_b    = (const float*)d_in[5];
  const float* w_W     = (const float*)d_in[6];
  const float* w_b     = (const float*)d_in[7];
  const float* v_W     = (const float*)d_in[8];
  const float* v_b     = (const float*)d_in[9];
  const float* gru_Wih = (const float*)d_in[10];
  const float* gru_Whh = (const float*)d_in[11];
  const float* gru_bih = (const float*)d_in[12];
  const float* gru_bhh = (const float*)d_in[13];
  const float* att_src = (const float*)d_in[14];
  const float* att_dst = (const float*)d_in[15];
  const float* gnn_W   = (const float*)d_in[16];
  const float* gnn_b   = (const float*)d_in[17];
  const float* out_W   = (const float*)d_in[18];
  const float* out_b   = (const float*)d_in[19];
  float* out = (float*)d_out;

  char* ws = (char*)d_ws;
  size_t off = 0;
  auto carve = [&](size_t bytes) -> void* {
    void* p = ws + off;
    off = (off + bytes + 255) & ~(size_t)255;
    return p;
  };
  float* hA       = (float*)carve(sizeof(float) * (size_t)NNODE * DH);
  float* hB       = (float*)carve(sizeof(float) * (size_t)NNODE * DH);
  float* gx       = (float*)carve(sizeof(float) * (size_t)NNODE * 3 * DH);
  float* att      = (float*)carve(sizeof(float) * (size_t)NNODE * DH);  // attended / tw
  float* gbuf     = (float*)carve(sizeof(float) * (size_t)NNODE * DH);  // g / tv
  float* agg      = (float*)carve(sizeof(float) * (size_t)NNODE * DH);
  _Float16* a16   = (_Float16*)carve(sizeof(_Float16) * (size_t)NNODE * DINP);
  _Float16* b16   = (_Float16*)carve(sizeof(_Float16) * (size_t)3 * DH * DINP);
  _Float16* w16   = (_Float16*)carve(sizeof(_Float16) * (size_t)DH * DH);
  _Float16* v16w  = (_Float16*)carve(sizeof(_Float16) * (size_t)DH * DH);
  float* logits   = (float*)carve(sizeof(float) * (size_t)NEDGE);
  float* eexp     = (float*)carve(sizeof(float) * (size_t)NEDGE);
  float* mN       = (float*)carve(sizeof(float) * NNODE);
  float* den      = (float*)carve(sizeof(float) * NNODE);
  float* deg      = (float*)carve(sizeof(float) * NNODE);
  float* sarr     = (float*)carve(sizeof(float) * NNODE);
  float* darr     = (float*)carve(sizeof(float) * NNODE);
  float* psum     = (float*)carve(sizeof(float) * NGRAPH * DH);
  float* pmax     = (float*)carve(sizeof(float) * NGRAPH * DH);
  float* pcnt     = (float*)carve(sizeof(float) * NGRAPH);

  const int TPB = 256;
  auto blocks1d = [](size_t n) { return (unsigned)((n + 255) / 256); };

  // ---- input projection: hA = x @ in_W^T + in_b (K padded 300 -> 320) ----
  wgm_cvt_f16<<<blocks1d((size_t)NNODE * DINP), TPB, 0, stream>>>(x, a16, NNODE, DIN, DINP);
  wgm_cvt_f16<<<blocks1d((size_t)DH * DINP), TPB, 0, stream>>>(in_W, b16, DH, DIN, DINP);
  wgm_gemm_wmma<<<dim3(NNODE / 128, DH / 64), TPB, 0, stream>>>(a16, b16, in_b, hA, NNODE, DH, DINP);

  // ---- SwiGLU weights (layer-invariant), convert once ----
  wgm_cvt_f16<<<blocks1d((size_t)DH * DH), TPB, 0, stream>>>(w_W, w16, DH, DH, DH);
  wgm_cvt_f16<<<blocks1d((size_t)DH * DH), TPB, 0, stream>>>(v_W, v16w, DH, DH, DH);

  // ---- GRU + SwiGLU stack ----
  for (int l = 0; l < NLAYER; ++l) {
    wgm_cvt_f16<<<blocks1d((size_t)NNODE * DH), TPB, 0, stream>>>(hA, a16, NNODE, DH, DH);
    wgm_cvt_f16<<<blocks1d((size_t)3 * DH * DH), TPB, 0, stream>>>(
        gru_Wih + (size_t)l * 3 * DH * DH, b16, 3 * DH, DH, DH);
    wgm_gemm_wmma<<<dim3(NNODE / 128, (3 * DH) / 64), TPB, 0, stream>>>(
        a16, b16, gru_bih + l * 3 * DH, gx, NNODE, 3 * DH, DH);
    wgm_gru_scan<<<1, 3 * DH, 0, stream>>>(
        gx, gru_Whh + (size_t)l * 3 * DH * DH, gru_bhh + l * 3 * DH, hB);
    wgm_cvt_f16<<<blocks1d((size_t)NNODE * DH), TPB, 0, stream>>>(hB, a16, NNODE, DH, DH);
    wgm_gemm_wmma<<<dim3(NNODE / 128, DH / 64), TPB, 0, stream>>>(a16, w16, w_b, att, NNODE, DH, DH);
    wgm_gemm_wmma<<<dim3(NNODE / 128, DH / 64), TPB, 0, stream>>>(a16, v16w, v_b, gbuf, NNODE, DH, DH);
    wgm_swiglu<<<blocks1d((size_t)NNODE * DH), TPB, 0, stream>>>(att, gbuf, hA, (size_t)NNODE * DH);
  }

  // ---- GAT attention + GNN message passing ----
  for (int l = 0; l < NLAYER; ++l) {
    wgm_attn_dots<<<NNODE / 8, TPB, 0, stream>>>(hA, att_src + l * DH, att_dst + l * DH, sarr, darr);
    wgm_fill<<<blocks1d(NNODE), TPB, 0, stream>>>(mN, NEG_INF, NNODE);
    wgm_fill<<<blocks1d(NNODE), TPB, 0, stream>>>(den, 0.0f, NNODE);
    wgm_fill<<<blocks1d(NNODE), TPB, 0, stream>>>(deg, 0.0f, NNODE);
    wgm_fill<<<blocks1d((size_t)NNODE * DH), TPB, 0, stream>>>(att, 0.0f, (size_t)NNODE * DH);
    wgm_fill<<<blocks1d((size_t)NNODE * DH), TPB, 0, stream>>>(agg, 0.0f, (size_t)NNODE * DH);
    wgm_edge_logit_max<<<NEDGE / 256, TPB, 0, stream>>>(ei, sarr, darr, logits, mN);
    wgm_edge_exp_den<<<NEDGE / 256, TPB, 0, stream>>>(ei, logits, mN, eexp, den);
    wgm_edge_attend<<<NEDGE / 8, TPB, 0, stream>>>(ei, ew, eexp, den, hA, att);
    wgm_cvt_f16<<<blocks1d((size_t)NNODE * DH), TPB, 0, stream>>>(att, a16, NNODE, DH, DH);
    wgm_cvt_f16<<<blocks1d((size_t)DH * DH), TPB, 0, stream>>>(gnn_W + (size_t)l * DH * DH, b16, DH, DH, DH);
    wgm_gemm_wmma<<<dim3(NNODE / 128, DH / 64), TPB, 0, stream>>>(a16, b16, gnn_b + l * DH, gbuf, NNODE, DH, DH);
    wgm_edge_gnn<<<NEDGE / 8, TPB, 0, stream>>>(ei, ew, gbuf, agg, deg);
    wgm_combine<<<blocks1d((size_t)NNODE * DH), TPB, 0, stream>>>(agg, deg, att, hA, (size_t)NNODE * DH);
  }

  // ---- readout ----
  wgm_fill<<<blocks1d((size_t)NGRAPH * DH), TPB, 0, stream>>>(psum, 0.0f, (size_t)NGRAPH * DH);
  wgm_fill<<<blocks1d((size_t)NGRAPH * DH), TPB, 0, stream>>>(pmax, NEG_INF, (size_t)NGRAPH * DH);
  wgm_fill<<<1, 64, 0, stream>>>(pcnt, 0.0f, (size_t)NGRAPH);
  wgm_pool<<<NNODE / 8, TPB, 0, stream>>>(hA, batch, psum, pmax, pcnt);
  wgm_readout<<<1, NGRAPH * DOUT, 0, stream>>>(psum, pmax, pcnt, out_W, out_b, out);
}
